// ChebyshevGNN_76175539962267
// MI455X (gfx1250) — matrix operand order repaired and checked by reading, not compiled, not run on hardware
//
#include <hip/hip_runtime.h>
#include <hip/hip_bf16.h>
#include <cstdint>

// ---------------------------------------------------------------------------
// Types for CDNA5 WMMA (wave32): v_wmma_f32_16x16x32_bf16
// ---------------------------------------------------------------------------
typedef __attribute__((ext_vector_type(16))) __bf16       v16bf;
typedef __attribute__((ext_vector_type(8)))  float        v8f;
typedef __attribute__((ext_vector_type(4)))  unsigned int u32x4;

union BF16Frag { v16bf v; u32x4 q[2]; };

__device__ inline unsigned short f2bf(float f) {
    unsigned u = __float_as_uint(f);
    u += 0x7FFFu + ((u >> 16) & 1u);          // round-to-nearest-even
    return (unsigned short)(u >> 16);
}

// ---------------------------------------------------------------------------
// Small utility kernels
// ---------------------------------------------------------------------------
__global__ void zero_f32_kernel(float* __restrict__ p, long n) {
    long i = (long)blockIdx.x * blockDim.x + threadIdx.x;
    if (i < n) p[i] = 0.0f;
}

__global__ void degree_kernel(const int* __restrict__ src, float* __restrict__ deg, int E) {
    int e = blockIdx.x * blockDim.x + threadIdx.x;
    if (e < E) atomicAdd(&deg[src[e]], 1.0f);
}

__global__ void dinv_kernel(float* __restrict__ deg, int N) {
    int i = blockIdx.x * blockDim.x + threadIdx.x;
    if (i < N) { float d = deg[i]; deg[i] = d > 0.0f ? rsqrtf(d) : 0.0f; }
}

__global__ void edgew_kernel(const int* __restrict__ src, const int* __restrict__ dst,
                             const float* __restrict__ dinv, float* __restrict__ w, int E) {
    int e = blockIdx.x * blockDim.x + threadIdx.x;
    if (e < E) w[e] = -dinv[src[e]] * dinv[dst[e]];
}

// ---------------------------------------------------------------------------
// Sparse propagation: out[dst] += w[e] * in[src]   (one wave32 per edge)
// ---------------------------------------------------------------------------
__global__ void prop_kernel(const float* __restrict__ in, const float* __restrict__ w,
                            const int* __restrict__ src, const int* __restrict__ dst,
                            float* __restrict__ out, int E, int D) {
    int tid  = blockIdx.x * blockDim.x + threadIdx.x;
    int e    = tid >> 5;
    int lane = tid & 31;
    if (e >= E) return;
    int   s  = src[e];
    int   d  = dst[e];
    float we = w[e];
    const float4* ip = (const float4*)(in + (size_t)s * D);
    float*        op = out + (size_t)d * D;
    for (int f = lane; f * 4 < D; f += 32) {
        float4 v = ip[f];
        atomicAdd(op + f * 4 + 0, we * v.x);
        atomicAdd(op + f * 4 + 1, we * v.y);
        atomicAdd(op + f * 4 + 2, we * v.z);
        atomicAdd(op + f * 4 + 3, we * v.w);
    }
}

// ---------------------------------------------------------------------------
// Build Tcat(bf16) = [T0 | T1 | T2] with T2 = 2*p2 - h, row-major N x 3D
// ---------------------------------------------------------------------------
__global__ void tcat_kernel(const float* __restrict__ h, const float* __restrict__ p1,
                            const float* __restrict__ p2, unsigned short* __restrict__ T,
                            int N, int D) {
    int idx = blockIdx.x * blockDim.x + threadIdx.x;
    if (idx >= N * D) return;
    int n = idx / D, c = idx % D;
    float hv = h[idx];
    float v1 = p1[idx];
    float v2 = 2.0f * p2[idx] - hv;
    size_t base = (size_t)n * (3 * D) + c;
    T[base]         = f2bf(hv);
    T[base + D]     = f2bf(v1);
    T[base + 2 * D] = f2bf(v2);
}

// ---------------------------------------------------------------------------
// Pack W (KD x F row-major f32 == the (3,Din,F) tensor flattened) into WMMA-B
// fragment order, bf16.  B order used here (same as A): lanes 0-15 -> K=2v,2v+1 ; lanes
// 16-31 -> K=16+2v,16+2v+1 ; N = lane&15.  Each lane owns 16 contiguous halfs.
// ---------------------------------------------------------------------------
__global__ void pack_w_kernel(const float* __restrict__ W, unsigned short* __restrict__ Wp,
                              int KD, int F) {
    int KT = KD / 32, NT = F / 16;
    int idx = blockIdx.x * blockDim.x + threadIdx.x;
    if (idx >= KT * NT * 32) return;
    int lane = idx & 31;
    int t    = idx >> 5;
    int nt   = t % NT;
    int kt   = t / NT;
    int hi   = lane >> 4, lo = lane & 15;
    int n    = nt * 16 + lo;
    size_t obase = ((size_t)(kt * NT + nt) * 32 + lane) * 16;
    for (int v = 0; v < 8; ++v) {
        int k0 = kt * 32 + hi * 16 + 2 * v;
        Wp[obase + 2 * v + 0] = f2bf(W[(size_t)k0 * F + n]);
        Wp[obase + 2 * v + 1] = f2bf(W[(size_t)(k0 + 1) * F + n]);
    }
}

// ---------------------------------------------------------------------------
// WMMA GEMM: out(N x F) = Tcat(N x KD, bf16) @ Wp(packed bf16) + bias
// One block handles 16 rows x F cols; one wave per 16-col tile.
// LN=true: fused bias + LayerNorm + ELU epilogue (F==128, 256 threads).
// ---------------------------------------------------------------------------
template <int KD, int F, bool LN>
__global__ __launch_bounds__((F / 16) * 32)
void gemm_wmma_kernel(const unsigned short* __restrict__ T,
                      const unsigned short* __restrict__ Wp,
                      const float* __restrict__ bias,
                      const float* __restrict__ g,
                      const float* __restrict__ lb,
                      float* __restrict__ out, int N) {
    constexpr int NT = F / 16;
    constexpr int KT = KD / 32;
    const int row0 = blockIdx.x * 16;
    const int wave = threadIdx.x >> 5;   // col tile index
    const int lane = threadIdx.x & 31;
    const int hi   = lane >> 4;
    const int lo   = lane & 15;

    int arow = row0 + lo;
    if (arow >= N) arow = N - 1;
    const unsigned short* aptr = T + (size_t)arow * KD;

    v8f acc = 0.0f;
#pragma unroll
    for (int kt = 0; kt < KT; ++kt) {
        BF16Frag a, b;
        // A 16x32 bf16 fragment: lanes 0-15 hold K0-15, lanes 16-31
        // hold K16-31  -> two contiguous 16B loads per lane.
        a.q[0] = *(const u32x4*)(aptr + kt * 32 + hi * 16);
        a.q[1] = *(const u32x4*)(aptr + kt * 32 + hi * 16 + 8);
        const u32x4* bp = (const u32x4*)(Wp + ((size_t)(kt * NT + wave) * 32 + lane) * 16);
        b.q[0] = bp[0];
        b.q[1] = bp[1];
        acc = __builtin_amdgcn_wmma_f32_16x16x32_bf16(false, a.v, false, b.v,
                                                      (short)0, acc, false, false);
    }

    if constexpr (LN) {
        // Spill 16 x F f32 tile to LDS, then fused bias + LayerNorm + ELU.
        __shared__ float smem[16][F + 4];
#pragma unroll
        for (int r = 0; r < 8; ++r)
            smem[hi * 8 + r][wave * 16 + lo] = acc[r];
        __syncthreads();

        const int row = threadIdx.x >> 4;    // 0..15
        const int sub = threadIdx.x & 15;    // 16 threads per row, 8 cols each
        float vals[8];
        float s = 0.0f, ss = 0.0f;
#pragma unroll
        for (int j = 0; j < 8; ++j) {
            int col = sub * 8 + j;
            float v = smem[row][col] + bias[col];
            vals[j] = v;
            s  += v;
            ss += v * v;
        }
        // reduce over the 16 lanes of this row (xor network stays in half-wave)
        for (int m = 1; m < 16; m <<= 1) {
            s  += __shfl_xor(s, m, 32);
            ss += __shfl_xor(ss, m, 32);
        }
        const float mean = s * (1.0f / F);
        const float var  = ss * (1.0f / F) - mean * mean;
        const float inv  = rsqrtf(var + 1e-5f);
        const int grow = row0 + row;
        if (grow < N) {
#pragma unroll
            for (int j = 0; j < 8; ++j) {
                int col = sub * 8 + j;
                float o = (vals[j] - mean) * inv * g[col] + lb[col];
                o = o > 0.0f ? o : expm1f(o);     // ELU(alpha=1)
                out[(size_t)grow * F + col] = o;
            }
        }
    } else {
        const int col = wave * 16 + lo;
        const float bv = bias[col];
#pragma unroll
        for (int r = 0; r < 8; ++r) {
            int grow = row0 + hi * 8 + r;
            if (grow < N) out[(size_t)grow * F + col] = acc[r] + bv;
        }
    }
}

// ---------------------------------------------------------------------------
// Host-side orchestration
// ---------------------------------------------------------------------------
static inline int cdiv(long a, long b) { return (int)((a + b - 1) / b); }

static void run_layer(const float* hin, int Din, const float* W, const float* bias,
                      const float* g, const float* lb, float* hout, bool last,
                      float* p1, float* p2, unsigned short* Tcat, unsigned short* Wp,
                      const float* w, const int* src, const int* dst,
                      int N, int E, hipStream_t stream) {
    const int KD = 3 * Din;
    const int F  = last ? 64 : 128;

    // T1 = L_hat @ h
    zero_f32_kernel<<<cdiv((long)N * Din, 256), 256, 0, stream>>>(p1, (long)N * Din);
    prop_kernel<<<cdiv((long)E * 32, 256), 256, 0, stream>>>(hin, w, src, dst, p1, E, Din);
    // p2 = L_hat @ T1   (T2 = 2*p2 - h computed in tcat)
    zero_f32_kernel<<<cdiv((long)N * Din, 256), 256, 0, stream>>>(p2, (long)N * Din);
    prop_kernel<<<cdiv((long)E * 32, 256), 256, 0, stream>>>(p1, w, src, dst, p2, E, Din);
    // Tcat = bf16([h | T1 | T2])
    tcat_kernel<<<cdiv((long)N * Din, 256), 256, 0, stream>>>(hin, p1, p2, Tcat, N, Din);
    // pack weights into WMMA-B fragment order
    pack_w_kernel<<<cdiv((long)(KD / 32) * (F / 16) * 32, 256), 256, 0, stream>>>(W, Wp, KD, F);

    const int blocks = cdiv(N, 16);
    if (Din == 64) {
        gemm_wmma_kernel<192, 128, true><<<blocks, 256, 0, stream>>>(Tcat, Wp, bias, g, lb, hout, N);
    } else if (!last) {
        gemm_wmma_kernel<384, 128, true><<<blocks, 256, 0, stream>>>(Tcat, Wp, bias, g, lb, hout, N);
    } else {
        gemm_wmma_kernel<384, 64, false><<<blocks, 128, 0, stream>>>(Tcat, Wp, bias, g, lb, hout, N);
    }
}

extern "C" void kernel_launch(void* const* d_in, const int* in_sizes, int n_in,
                              void* d_out, int out_size, void* d_ws, size_t ws_size,
                              hipStream_t stream) {
    const float* x   = (const float*)d_in[0];
    const int*   ei  = (const int*)d_in[1];
    const float* W0  = (const float*)d_in[2];
    const float* b0  = (const float*)d_in[3];
    const float* W1  = (const float*)d_in[4];
    const float* b1  = (const float*)d_in[5];
    const float* W2  = (const float*)d_in[6];
    const float* b2  = (const float*)d_in[7];
    const float* W3  = (const float*)d_in[8];
    const float* b3  = (const float*)d_in[9];
    const float* lng = (const float*)d_in[10];
    const float* lnb = (const float*)d_in[11];

    const int N = in_sizes[0] / 64;
    const int E = in_sizes[1] / 2;
    const int* src = ei;
    const int* dst = ei + E;

    // carve workspace (256B aligned chunks)
    char* p = (char*)d_ws;
    auto alloc = [&](size_t bytes) -> char* {
        char* r = p;
        p += (bytes + 255) & ~(size_t)255;
        return r;
    };
    float*          dinv = (float*)alloc((size_t)N * 4);
    float*          w    = (float*)alloc((size_t)E * 4);
    float*          h    = (float*)alloc((size_t)N * 128 * 4);
    float*          p1   = (float*)alloc((size_t)N * 128 * 4);
    float*          p2   = (float*)alloc((size_t)N * 128 * 4);
    unsigned short* Tcat = (unsigned short*)alloc((size_t)N * 384 * 2);
    unsigned short* Wp   = (unsigned short*)alloc((size_t)384 * 128 * 2);

    // symmetric-normalized scaled-Laplacian edge weights
    zero_f32_kernel<<<cdiv(N, 256), 256, 0, stream>>>(dinv, N);
    degree_kernel<<<cdiv(E, 256), 256, 0, stream>>>(src, dinv, E);
    dinv_kernel<<<cdiv(N, 256), 256, 0, stream>>>(dinv, N);
    edgew_kernel<<<cdiv(E, 256), 256, 0, stream>>>(src, dst, dinv, w, E);

    // layer 0: 64 -> 128 (LN+ELU), input is x (never mutated)
    run_layer(x, 64, W0, b0, lng + 0, lnb + 0, h, false,
              p1, p2, Tcat, Wp, w, src, dst, N, E, stream);
    // layer 1: 128 -> 128
    run_layer(h, 128, W1, b1, lng + 128, lnb + 128, h, false,
              p1, p2, Tcat, Wp, w, src, dst, N, E, stream);
    // layer 2: 128 -> 128
    run_layer(h, 128, W2, b2, lng + 256, lnb + 256, h, false,
              p1, p2, Tcat, Wp, w, src, dst, N, E, stream);
    // layer 3: 128 -> 64, no LN/ELU, straight to d_out
    run_layer(h, 128, W3, b3, nullptr, nullptr, (float*)d_out, true,
              p1, p2, Tcat, Wp, w, src, dst, N, E, stream);
}
